// AttentionHead_7834020348407
// MI455X (gfx1250) — matrix operand order repaired and checked by reading, not compiled, not run on hardware
//
#include <hip/hip_runtime.h>

typedef __attribute__((ext_vector_type(16))) _Float16 v16h;
typedef __attribute__((ext_vector_type(8)))  _Float16 h8v;
typedef __attribute__((ext_vector_type(4)))  _Float16 h4v;
typedef __attribute__((ext_vector_type(8)))  float    v8f;
typedef __attribute__((ext_vector_type(4)))  float    f4v;

#define NB 4
#define NS 4096
#define NE 1024
#define NH 64
#define NROWS (NB * NS)   // 16384
#define WELEMS (NH * NE)  // 65536 per weight matrix

static __device__ __forceinline__ v8f wmma_f16(v16h a, v16h b, v8f c) {
  // D = A(16x32 f16) * B(32x16 f16) + C(16x16 f32)
  return __builtin_amdgcn_wmma_f32_16x16x32_f16(
      /*neg_a=*/false, a, /*neg_b=*/false, b,
      /*c_mod=*/(short)0, c, /*reuse_a=*/false, /*reuse_b=*/false);
}

// A fragment from an f32 row: halves 0..15 = row[off..off+15]
static __device__ __forceinline__ v16h cvtA_f32(const float* __restrict__ row, int off) {
  f4v a = *(const f4v*)(row + off);
  f4v b = *(const f4v*)(row + off + 4);
  f4v c = *(const f4v*)(row + off + 8);
  f4v d = *(const f4v*)(row + off + 12);
  v16h r;
  r[0]=(_Float16)a[0]; r[1]=(_Float16)a[1]; r[2]=(_Float16)a[2]; r[3]=(_Float16)a[3];
  r[4]=(_Float16)b[0]; r[5]=(_Float16)b[1]; r[6]=(_Float16)b[2]; r[7]=(_Float16)b[3];
  r[8]=(_Float16)c[0]; r[9]=(_Float16)c[1]; r[10]=(_Float16)c[2]; r[11]=(_Float16)c[3];
  r[12]=(_Float16)d[0]; r[13]=(_Float16)d[1]; r[14]=(_Float16)d[2]; r[15]=(_Float16)d[3];
  return r;
}

// A fragment from an f16 row: halves 0..15 = row[off..off+15]
static __device__ __forceinline__ v16h load_a16(const _Float16* __restrict__ row, int off) {
  h8v a = *(const h8v*)(row + off);
  h8v b = *(const h8v*)(row + off + 8);
  v16h r;
  r[0]=a[0]; r[1]=a[1]; r[2]=a[2]; r[3]=a[3]; r[4]=a[4]; r[5]=a[5]; r[6]=a[6]; r[7]=a[7];
  r[8]=b[0]; r[9]=b[1]; r[10]=b[2]; r[11]=b[3]; r[12]=b[4]; r[13]=b[5]; r[14]=b[6]; r[15]=b[7];
  return r;
}

// B fragment: 16 contiguous f16
static __device__ __forceinline__ v16h load_b16(const _Float16* __restrict__ p) {
  h8v a = *(const h8v*)(p);
  h8v b = *(const h8v*)(p + 8);
  v16h r;
  r[0]=a[0]; r[1]=a[1]; r[2]=a[2]; r[3]=a[3]; r[4]=a[4]; r[5]=a[5]; r[6]=a[6]; r[7]=a[7];
  r[8]=b[0]; r[9]=b[1]; r[10]=b[2]; r[11]=b[3]; r[12]=b[4]; r[13]=b[5]; r[14]=b[6]; r[15]=b[7];
  return r;
}

// ---------------------------------------------------------------------------
// Kernel 0: one-shot f32 -> f16 conversion of Wq/Wk/Wv (256 KB each -> 128 KB).
// Removes all B-side converts from the hot GEMM loop.
// ---------------------------------------------------------------------------
__global__ __launch_bounds__(256) void wcvt_kernel(
    const float* __restrict__ Wq, const float* __restrict__ Wk,
    const float* __restrict__ Wv,
    _Float16* __restrict__ wq, _Float16* __restrict__ wk,
    _Float16* __restrict__ wv) {
  const int i = (blockIdx.x * 256 + threadIdx.x) * 4;   // 4 elements per thread
  if (i < WELEMS) {
    f4v q = *(const f4v*)(Wq + i);
    f4v k = *(const f4v*)(Wk + i);
    f4v v = *(const f4v*)(Wv + i);
    h4v qh, kh, vh;
    qh[0]=(_Float16)q[0]; qh[1]=(_Float16)q[1]; qh[2]=(_Float16)q[2]; qh[3]=(_Float16)q[3];
    kh[0]=(_Float16)k[0]; kh[1]=(_Float16)k[1]; kh[2]=(_Float16)k[2]; kh[3]=(_Float16)k[3];
    vh[0]=(_Float16)v[0]; vh[1]=(_Float16)v[1]; vh[2]=(_Float16)v[2]; vh[3]=(_Float16)v[3];
    *(h4v*)(wq + i) = qh;
    *(h4v*)(wk + i) = kh;
    *(h4v*)(wv + i) = vh;
  }
}

// ---------------------------------------------------------------------------
// Kernel 1: q/k/v projections.  y = x @ W^T, output f16 row-major [NROWS, 64].
// One 16-row M-tile per wave, N=64 (4 tiles), K=1024 in chunks of 32.
// The softmax scale H^-0.5 = 0.125 is folded into q at store time.
// ---------------------------------------------------------------------------
__global__ __launch_bounds__(256) void qkv_proj_kernel(
    const float* __restrict__ x,
    const _Float16* __restrict__ wq, const _Float16* __restrict__ wk,
    const _Float16* __restrict__ wv,
    _Float16* __restrict__ q16, _Float16* __restrict__ k16,
    _Float16* __restrict__ v16) {
  const int lane = threadIdx.x & 31;
  const int wave = threadIdx.x >> 5;
  const int hi   = lane >> 4;      // 0: lanes 0-15, 1: lanes 16-31
  const int l16  = lane & 15;
  const int m0   = (blockIdx.x * 8 + wave) * 16;   // first row of this M tile

  v8f zero = {};
  v8f aq[4], ak[4], av[4];
#pragma unroll
  for (int t = 0; t < 4; ++t) { aq[t] = zero; ak[t] = zero; av[t] = zero; }

  const float* xrow = x + (size_t)(m0 + l16) * NE;   // A: lane row = m0 + l16

  for (int kc = 0; kc < NE; kc += 32) {
    v16h a = cvtA_f32(xrow, kc + hi * 16);
    const int kb = kc + hi * 16;
#pragma unroll
    for (int t = 0; t < 4; ++t) {
      const size_t n = (size_t)(t * 16 + l16);       // B: lane col = n
      v16h bq = load_b16(wq + n * NE + kb);
      v16h bk = load_b16(wk + n * NE + kb);
      v16h bv = load_b16(wv + n * NE + kb);
      aq[t] = wmma_f16(a, bq, aq[t]);
      ak[t] = wmma_f16(a, bk, ak[t]);
      av[t] = wmma_f16(a, bv, av[t]);
    }
  }

  // D layout: lane holds (row = r + hi*8, col = t*16 + l16) in VGPR r
#pragma unroll
  for (int t = 0; t < 4; ++t) {
#pragma unroll
    for (int r = 0; r < 8; ++r) {
      const size_t idx = (size_t)(m0 + r + hi * 8) * NH + t * 16 + l16;
      q16[idx] = (_Float16)(aq[t][r] * 0.125f);   // fold H^-0.5 into q
      k16[idx] = (_Float16)ak[t][r];
      v16[idx] = (_Float16)av[t][r];
    }
  }
}

// ---------------------------------------------------------------------------
// Kernel 2: causal flash attention. One 16-row Q tile per wave, KV blocks of 32.
// ---------------------------------------------------------------------------
__global__ __launch_bounds__(256) void flash_attn_kernel(
    const _Float16* __restrict__ q16, const _Float16* __restrict__ k16,
    const _Float16* __restrict__ v16, float* __restrict__ out) {
  __shared__ _Float16 pbuf[8][16][40];   // per-wave 16x32 P tile (+pad)

  const int lane = threadIdx.x & 31;
  const int wave = threadIdx.x >> 5;
  const int hi   = lane >> 4;
  const int l16  = lane & 15;
  const int qt   = blockIdx.x * 8 + wave;     // 0..1023 query tiles
  const int b    = qt >> 8;                   // 256 tiles per batch
  const int ql   = qt & 255;
  const int qbase = ql * 16;
  const size_t base = (size_t)b * NS * NH;

  // Q as A operand: two 16x32 fragments covering head dims 0..31 and 32..63
  const _Float16* qrow = q16 + base + (size_t)(qbase + l16) * NH;
  const v16h aq0 = load_a16(qrow, hi * 16);
  const v16h aq1 = load_a16(qrow + 32, hi * 16);

  v8f zero = {};
  v8f o0 = zero, o1 = zero, o2 = zero, o3 = zero;
  float M[8], L[8];
#pragma unroll
  for (int r = 0; r < 8; ++r) { M[r] = -1e30f; L[r] = 0.0f; }

  const int jmax = ql >> 1;     // last KV block (of 32) touching the causal region

  for (int j = 0; j <= jmax; ++j) {
    // ---- S = Q * K^T for 32 keys (two 16-col tiles), f32 accumulate ----
    const _Float16* kr0 = k16 + base + (size_t)(j * 32 + l16) * NH;
    const _Float16* kr1 = kr0 + 16 * NH;
    v8f s0 = zero, s1 = zero;
    s0 = wmma_f16(aq0, load_b16(kr0 + hi * 16), s0);
    s0 = wmma_f16(aq1, load_b16(kr0 + 32 + hi * 16), s0);
    s1 = wmma_f16(aq0, load_b16(kr1 + hi * 16), s1);
    s1 = wmma_f16(aq1, load_b16(kr1 + 32 + hi * 16), s1);

    __builtin_prefetch(kr0 + 32 * NH, 0, 1);   // next K block -> global_prefetch_b8

    // ---- online softmax (rows live across 16 lanes of each half-wave) ----
    float p0[8], p1[8];
#pragma unroll
    for (int r = 0; r < 8; ++r) {
      const int row = qbase + r + hi * 8;
      const int c0  = j * 32 + l16;
      float v0 = (c0 <= row)      ? s0[r] : -1e30f;  // scale pre-folded into q
      float v1 = (c0 + 16 <= row) ? s1[r] : -1e30f;
      float mx = fmaxf(v0, v1);
      mx = fmaxf(mx, __shfl_xor(mx, 1, 32));
      mx = fmaxf(mx, __shfl_xor(mx, 2, 32));
      mx = fmaxf(mx, __shfl_xor(mx, 4, 32));
      mx = fmaxf(mx, __shfl_xor(mx, 8, 32));
      mx = fmaxf(mx, M[r]);
      const float f  = __expf(M[r] - mx);
      const float e0 = __expf(v0 - mx);
      const float e1 = __expf(v1 - mx);
      float sum = e0 + e1;
      sum += __shfl_xor(sum, 1, 32);
      sum += __shfl_xor(sum, 2, 32);
      sum += __shfl_xor(sum, 4, 32);
      sum += __shfl_xor(sum, 8, 32);
      M[r] = mx;
      L[r] = L[r] * f + sum;
      p0[r] = e0; p1[r] = e1;
      o0[r] *= f; o1[r] *= f; o2[r] *= f; o3[r] *= f;
    }

    // ---- P: D-layout -> A-layout via per-wave LDS staging ----
#pragma unroll
    for (int r = 0; r < 8; ++r) {
      pbuf[wave][r + hi * 8][l16]      = (_Float16)p0[r];
      pbuf[wave][r + hi * 8][l16 + 16] = (_Float16)p1[r];
    }
    asm volatile("s_wait_dscnt 0" ::: "memory");
    const v16h ap = load_a16(&pbuf[wave][l16][0], hi * 16);

    // ---- V as B operand (32 keys x 64 head dims, 4 N tiles) ----
    const _Float16* vp = v16 + base + (size_t)(j * 32 + hi * 16) * NH + l16;
    __builtin_prefetch(vp + 32 * NH, 0, 1);    // next V block
    v16h bv0, bv1, bv2, bv3;
#pragma unroll
    for (int h = 0; h < 16; ++h) {
      const _Float16* vr = vp + (size_t)h * NH;
      bv0[h] = vr[0];
      bv1[h] = vr[16];
      bv2[h] = vr[32];
      bv3[h] = vr[48];
    }
    o0 = wmma_f16(ap, bv0, o0);
    o1 = wmma_f16(ap, bv1, o1);
    o2 = wmma_f16(ap, bv2, o2);
    o3 = wmma_f16(ap, bv3, o3);
  }

  // ---- normalize and store f32 output [B,S,H] ----
  float* orow = out + base + (size_t)qbase * NH;
#pragma unroll
  for (int r = 0; r < 8; ++r) {
    const float inv = 1.0f / L[r];
    const size_t rr = (size_t)(r + hi * 8) * NH + l16;
    orow[rr]      = o0[r] * inv;
    orow[rr + 16] = o1[r] * inv;
    orow[rr + 32] = o2[r] * inv;
    orow[rr + 48] = o3[r] * inv;
  }
}

// ---------------------------------------------------------------------------
extern "C" void kernel_launch(void* const* d_in, const int* in_sizes, int n_in,
                              void* d_out, int out_size, void* d_ws, size_t ws_size,
                              hipStream_t stream) {
  const float* x  = (const float*)d_in[0];
  const float* Wq = (const float*)d_in[1];
  const float* Wk = (const float*)d_in[2];
  const float* Wv = (const float*)d_in[3];
  float* out = (float*)d_out;

  // workspace: q16 | k16 | v16 (2 MB each) | wq16 | wk16 | wv16 (128 KB each)
  _Float16* q16  = (_Float16*)d_ws;
  _Float16* k16  = q16 + (size_t)NROWS * NH;
  _Float16* v16  = k16 + (size_t)NROWS * NH;
  _Float16* wq16 = v16 + (size_t)NROWS * NH;
  _Float16* wk16 = wq16 + WELEMS;
  _Float16* wv16 = wk16 + WELEMS;

  // 65536 elems / 4 per thread / 256 threads = 64 blocks
  wcvt_kernel<<<64, 256, 0, stream>>>(Wq, Wk, Wv, wq16, wk16, wv16);
  // 16384 rows / 16 per wave / 8 waves per block = 128 blocks
  qkv_proj_kernel<<<128, 256, 0, stream>>>(x, wq16, wk16, wv16, q16, k16, v16);
  // 1024 query tiles / 8 waves per block = 128 blocks
  flash_attn_kernel<<<128, 256, 0, stream>>>(q16, k16, v16, out);
}
